// Quantization_137438953784
// MI455X (gfx1250) — compile-verified, matrix operand order checked by reading, not run on hardware
//
#include <hip/hip_runtime.h>
#include <hip/hip_bf16.h>
#include <stdint.h>

// Problem constants (must match reference)
#define NUM_EMBED 1024
#define LATENT    256
#define BSZ       131072
#define BETA      0.25f

// Tiling
#define M_TILE 128               // z rows per workgroup (8 waves x 16 rows)
#define CHUNK  64                // codebook rows staged in LDS per step (32 KB bf16)
#define NCHUNK (NUM_EMBED / CHUNK)
#define NTILES (CHUNK / 16)      // 4 N-tiles per chunk
#define KSTEPS (LATENT / 32)     // 8 WMMA K-steps to cover D=256
#define N16    ((CHUNK * LATENT * 2) / 16)   // 2048 x 16B transfers per chunk
#define AIT    (N16 / 256)       // async ops issued per wave per chunk (= 8)

typedef __attribute__((ext_vector_type(16))) __bf16 v16bf;
typedef __attribute__((ext_vector_type(8)))  float  v8f;
typedef __attribute__((ext_vector_type(4)))  int    v4i;

// Address-space-qualified pointer types for the CDNA5 async-to-LDS builtin
// (probe-confirmed signature: (int4 global*, int4 lds*, imm offset, imm cpol)).
typedef __attribute__((address_space(1))) v4i gv4i;   // global
typedef __attribute__((address_space(3))) v4i lv4i;   // LDS

#define HAVE_ASYNC 0
#if defined(__has_builtin)
#if __has_builtin(__builtin_amdgcn_global_load_async_to_lds_b128) && \
    __has_builtin(__builtin_amdgcn_s_wait_asynccnt)
#undef HAVE_ASYNC
#define HAVE_ASYNC 1
#endif
#endif

// ---------------------------------------------------------------------------
// Kernel 1: codebook f32 -> bf16, per-row squared norms, zero the loss scalar.
// ---------------------------------------------------------------------------
__global__ __launch_bounds__(LATENT) void vq_prep(
    const float* __restrict__ cb, __bf16* __restrict__ cb_bf,
    float* __restrict__ cnorm, float* __restrict__ loss_out)
{
    const int k = blockIdx.x;
    const int t = threadIdx.x;
    const float v = cb[(size_t)k * LATENT + t];
    cb_bf[(size_t)k * LATENT + t] = (__bf16)v;

    float s = v * v;
    #pragma unroll
    for (int off = 16; off > 0; off >>= 1) s += __shfl_xor(s, off, 32);

    __shared__ float red[LATENT / 32];
    if ((t & 31) == 0) red[t >> 5] = s;
    __syncthreads();
    if (t == 0) {
        float tot = 0.f;
        #pragma unroll
        for (int i = 0; i < LATENT / 32; ++i) tot += red[i];
        cnorm[k] = tot;
        if (k == 0) *loss_out = 0.f;   // re-zeroed every launch (graph-replay safe)
    }
}

// ---------------------------------------------------------------------------
// Kernel 2: argmin_k ( ||c_k||^2 - 2 z.c_k )  via bf16 WMMA, f32 accumulate.
// Double-buffered async LDS staging + batched B-fragments so the 8 WMMAs of a
// tile issue back-to-back instead of stalling on per-step s_wait_dscnt.
// ---------------------------------------------------------------------------
__global__ __launch_bounds__(256) void vq_argmin(
    const float*  __restrict__ z,
    const __bf16* __restrict__ cb_bf,
    const float*  __restrict__ cnorm,
    int*          __restrict__ ids_out)
{
    __shared__ __bf16 lds_cb[2][CHUNK * LATENT];   // 2 x 32 KB (WGP has 320 KB)

    const int tid  = threadIdx.x;
    const int lane = tid & 31;
    const int wave = tid >> 5;
    const int half = lane >> 4;     // 0: lanes 0-15, 1: lanes 16-31
    const int l16  = lane & 15;

    const int rowBase = blockIdx.x * M_TILE + wave * 16;
    const int myRow   = rowBase + l16;

    // --- Load this wave's z rows once; convert to bf16 A-fragments (resident).
    // 16-bit A 16x32 layout: lanes 0-15 hold K={k0*32+0..7, k0*32+16..23},
    // lanes 16-31 hold K shifted by +8.
    v16bf a[KSTEPS];
    {
        const float* zr = z + (size_t)myRow * LATENT;
        #pragma unroll
        for (int k0 = 0; k0 < KSTEPS; ++k0) {
            const float* p0 = zr + k0 * 32 + half * 8;
            v16bf t;
            #pragma unroll
            for (int j = 0; j < 8; ++j) t[j]     = (__bf16)p0[j];
            #pragma unroll
            for (int j = 0; j < 8; ++j) t[8 + j] = (__bf16)p0[16 + j];
            a[k0] = t;
        }
    }

    float bestv[8];
    int   besti[8];
    #pragma unroll
    for (int r = 0; r < 8; ++r) { bestv[r] = 3.4e38f; besti[r] = 0; }

#if HAVE_ASYNC
    // Prologue: prefetch chunk 0 into buffer 0.
    {
        gv4i* gsrc = (gv4i*)cb_bf;
        lv4i* ldst = (lv4i*)&lds_cb[0][0];
        for (int i = tid; i < N16; i += 256)
            __builtin_amdgcn_global_load_async_to_lds_b128(gsrc + i, ldst + i, 0, 0);
    }
#endif

    for (int chunk = 0; chunk < NCHUNK; ++chunk) {
        const int cur = chunk & 1;

#if HAVE_ASYNC
        // Prefetch chunk+1 into the other buffer, then retire only the current
        // chunk's 8 in-order transfers (ASYNCcnt <= AIT keeps prefetch in flight).
        if (chunk + 1 < NCHUNK) {
            gv4i* gsrc = (gv4i*)(cb_bf + (size_t)(chunk + 1) * CHUNK * LATENT);
            lv4i* ldst = (lv4i*)&lds_cb[cur ^ 1][0];
            for (int i = tid; i < N16; i += 256)
                __builtin_amdgcn_global_load_async_to_lds_b128(gsrc + i, ldst + i, 0, 0);
            __builtin_amdgcn_s_wait_asynccnt(AIT);
        } else {
            __builtin_amdgcn_s_wait_asynccnt(0);
        }
#else
        __syncthreads();
        {
            const uint4* s4 = (const uint4*)(cb_bf + (size_t)chunk * CHUNK * LATENT);
            uint4*       d4 = (uint4*)&lds_cb[cur][0];
            for (int i = tid; i < (CHUNK * LATENT) / 8; i += 256) d4[i] = s4[i];
        }
#endif
        __syncthreads();   // current buffer fully resident for all waves

        const __bf16* buf = &lds_cb[cur][0];

        #pragma unroll
        for (int t = 0; t < NTILES; ++t) {
            const int colInChunk = t * 16 + l16;
            const int codeId     = chunk * CHUNK + colInChunk;
            const float cn       = cnorm[codeId];

            // --- Batch-load all 8 B-fragments into distinct registers first
            // (two 16B ds loads each), so the WMMAs below issue back-to-back.
            union frag { uint4 u[2]; v16bf v; };
            frag bf[KSTEPS];
            #pragma unroll
            for (int k0 = 0; k0 < KSTEPS; ++k0) {
                const __bf16* bp = buf + colInChunk * LATENT + k0 * 32 + half * 8;
                bf[k0].u[0] = *(const uint4*)bp;
                bf[k0].u[1] = *(const uint4*)(bp + 16);
            }

            v8f acc = {};
            #pragma unroll
            for (int k0 = 0; k0 < KSTEPS; ++k0)
                acc = __builtin_amdgcn_wmma_f32_16x16x32_bf16(
                          false, a[k0], false, bf[k0].v, (short)0, acc, false, false);

            // dist - ||z||^2 = cn - 2*dot  (row norm irrelevant to argmin).
            // Strict '<' keeps the first (lowest) index on ties.
            #pragma unroll
            for (int r = 0; r < 8; ++r) {
                const float dv = cn - 2.0f * acc[r];
                if (dv < bestv[r]) { bestv[r] = dv; besti[r] = codeId; }
            }
        }
        __syncthreads();   // all reads of this buffer done before it is refilled
    }

    // --- Cross-lane argmin: the 16 lanes of each half share the same 8 rows.
    // xor masks {1,2,4,8} never cross the half boundary in wave32.
    #pragma unroll
    for (int r = 0; r < 8; ++r) {
        float v = bestv[r]; int i = besti[r];
        #pragma unroll
        for (int m = 1; m < 16; m <<= 1) {
            const float ov = __shfl_xor(v, m, 32);
            const int   oi = __shfl_xor(i, m, 32);
            if (ov < v || (ov == v && oi < i)) { v = ov; i = oi; }
        }
        bestv[r] = v; besti[r] = i;
    }

    if (l16 == 0) {
        #pragma unroll
        for (int r = 0; r < 8; ++r)
            ids_out[rowBase + half * 8 + r] = besti[r];
    }
}

// ---------------------------------------------------------------------------
// Kernel 3: gather codebook rows (exact f32) as the STE output + loss.
// ---------------------------------------------------------------------------
__global__ __launch_bounds__(LATENT) void vq_gather(
    const float* __restrict__ z, const float* __restrict__ cb,
    const int*   __restrict__ ids, float* __restrict__ out,
    float*       __restrict__ loss)
{
    const int row = blockIdx.x;
    const int t   = threadIdx.x;
    const int id  = ids[row];

    const float cq = cb[(size_t)id  * LATENT + t];
    const float zv = z [(size_t)row * LATENT + t];
    out[(size_t)row * LATENT + t] = cq;      // z + sg(z_q - z) == z_q forward

    const float d = cq - zv;
    float s = d * d;
    #pragma unroll
    for (int off = 16; off > 0; off >>= 1) s += __shfl_xor(s, off, 32);

    __shared__ float red[LATENT / 32];
    if ((t & 31) == 0) red[t >> 5] = s;
    __syncthreads();
    if (t == 0) {
        float tot = 0.f;
        #pragma unroll
        for (int i = 0; i < LATENT / 32; ++i) tot += red[i];
        atomicAdd(loss, tot * ((1.0f + BETA) / ((float)BSZ * (float)LATENT)));
    }
}

// ---------------------------------------------------------------------------
// Launch. d_in[0] = z [B,1,D] f32, d_in[1] = codebook [K,D] f32.
// d_out = [ z_q_ste (B*D f32) | loss (1 f32) | ids (B i32, bit-cast) ].
// d_ws  = [ bf16 codebook 512 KB | cnorm 4 KB ].
// ---------------------------------------------------------------------------
extern "C" void kernel_launch(void* const* d_in, const int* in_sizes, int n_in,
                              void* d_out, int out_size, void* d_ws, size_t ws_size,
                              hipStream_t stream)
{
    (void)in_sizes; (void)n_in; (void)out_size; (void)ws_size;

    const float* z  = (const float*)d_in[0];
    const float* cb = (const float*)d_in[1];

    __bf16* cb_bf = (__bf16*)d_ws;
    float*  cnorm = (float*)((char*)d_ws + (size_t)NUM_EMBED * LATENT * sizeof(__bf16));

    float* out  = (float*)d_out;
    float* loss = out + (size_t)BSZ * LATENT;
    int*   ids  = (int*)(loss + 1);

    vq_prep  <<<NUM_EMBED,    LATENT, 0, stream>>>(cb, cb_bf, cnorm, loss);
    vq_argmin<<<BSZ / M_TILE, 256,    0, stream>>>(z, cb_bf, cnorm, ids);
    vq_gather<<<BSZ,          LATENT, 0, stream>>>(z, cb, ids, out, loss);
}
